// GCNAttn_1494648619259
// MI455X (gfx1250) — compile-verified, matrix operand order checked by reading, not run on hardware
//
#include <hip/hip_runtime.h>

#define HH 2
#define DW 128          // H * C (both layers)
#define FIN 32
#define BB 2
#define NEG_SLOPE 0.1f

typedef __attribute__((ext_vector_type(16))) _Float16 v16h;
typedef __attribute__((ext_vector_type(8)))  _Float16 v8h;
typedef __attribute__((ext_vector_type(8)))  float    v8f;

// ---------------- utility kernels ----------------
__global__ void fill_kernel(float* p, float v, int n) {
  int i = blockIdx.x * blockDim.x + threadIdx.x;
  if (i < n) p[i] = v;
}

// x [B, FIN, N] fp32 -> featH [(b*N+n)*FIN + k] f16
__global__ void pack_feat_kernel(const float* __restrict__ x, _Float16* __restrict__ featH, int N) {
  int i = blockIdx.x * blockDim.x + threadIdx.x;  // over B*N*FIN
  if (i >= BB * N * FIN) return;
  int k = i & (FIN - 1);
  int n = (i >> 5) % N;
  int b = (i >> 5) / N;
  featH[i] = (_Float16)x[((size_t)b * FIN + k) * N + n];
}

// ---------------- WMMA GEMM: out[rows16,128] = A[rows16,K](f16) @ W[K,128] + bias -------------
// rows16 is a multiple of 16 (buffers padded in workspace): no guards, full tiles only.
// W is staged into LDS pre-swizzled into the exact wave32 B-fragment layout:
// sW[((t*NC + c)*32 + lane)*16 + i] = W[k, n]  with
//   g = lane>>4, n = t*16 + (lane&15),
//   k = c*32 + (i<8 ? g*8+i : 16 + g*8 + (i-8))
// so each B fragment is two contiguous ds_load_b128 per lane.
template <int K>
__global__ void __launch_bounds__(256)
gemm_wmma_kernel(const _Float16* __restrict__ A, const float* __restrict__ W,
                 const float* __restrict__ bias, float* __restrict__ out, int rows16) {
  constexpr int NC = K / 32;      // 32-wide K chunks
  __shared__ _Float16 sW[K * DW];

  for (int idx = threadIdx.x; idx < K * DW; idx += 256) {
    const int i  = idx & 15;
    const int l  = (idx >> 4) & 31;
    const int c  = (idx >> 9) % NC;
    const int t  = (idx >> 9) / NC;
    const int gg = l >> 4;
    const int n  = t * 16 + (l & 15);
    const int k  = c * 32 + (i < 8 ? gg * 8 + i : 16 + gg * 8 + (i - 8));
    sW[idx] = (_Float16)W[k * DW + n];
  }
  __syncthreads();

  const int wave = threadIdx.x >> 5;
  const int lane = threadIdx.x & 31;
  const int row0 = (blockIdx.x * 8 + wave) * 16;
  if (row0 >= rows16) return;

  const int g    = lane >> 4;     // K-group within lane pair
  const int mloc = lane & 15;     // M (A) or N (B/D) within tile
  const _Float16* Ar = A + (size_t)(row0 + mloc) * K;

  v16h afrag[NC];
  for (int c = 0; c < NC; ++c) {
    v8h lo = *(const v8h*)(Ar + c * 32 + g * 8);        // K = c*32 + g*8 .. +7
    v8h hi = *(const v8h*)(Ar + c * 32 + 16 + g * 8);   // K = c*32 + 16 + g*8 .. +7
    afrag[c] = __builtin_shufflevector(lo, hi, 0, 1, 2, 3, 4, 5, 6, 7,
                                       8, 9, 10, 11, 12, 13, 14, 15);
  }

  for (int t = 0; t < 8; ++t) {                          // 8 output col-tiles of 16
    const int n = t * 16 + mloc;
    v8f acc = {};
    for (int c = 0; c < NC; ++c) {
      const _Float16* bp = &sW[((t * NC + c) * 32 + lane) * 16];
      v8h blo = *(const v8h*)bp;                         // ds_load_b128
      v8h bhi = *(const v8h*)(bp + 8);                   // ds_load_b128
      v16h bfrag = __builtin_shufflevector(blo, bhi, 0, 1, 2, 3, 4, 5, 6, 7,
                                           8, 9, 10, 11, 12, 13, 14, 15);
      acc = __builtin_amdgcn_wmma_f32_16x16x32_f16(false, afrag[c], false, bfrag,
                                                   (short)0, acc, false, false);
    }
    const float bv = bias[n];
    float* op = out + (size_t)row0 * DW + n;             // D: M = r + 8*g, N = mloc
    for (int r = 0; r < 8; ++r)
      op[(size_t)(r + 8 * g) * DW] = acc[r] + bv;
  }
}

// ---------------- edge kernels ----------------
__device__ inline void atomicMaxF32(float* a, float v) {
  if (v >= 0.0f) atomicMax((int*)a, __float_as_int(v));
  else           atomicMin((unsigned int*)a, (unsigned int)__float_as_int(v));
}

__device__ inline void edge_pair(const int* ei, int e, int E, int& src, int& dst) {
  if (e < E) { src = ei[e]; dst = ei[E + e]; }
  else       { src = dst = e - E; }          // self-loops appended
}

// one wave per (edge, batch): scores + segment max
__global__ void __launch_bounds__(256)
edge_score_kernel(const float* __restrict__ xl, const float* __restrict__ xr,
                  const int* __restrict__ ei, const float* __restrict__ att,
                  float* __restrict__ scores, float* __restrict__ mbuf,
                  int E, int Etot, int N) {
  const int e    = (blockIdx.x * 256 + threadIdx.x) >> 5;
  const int lane = threadIdx.x & 31;
  const int b    = blockIdx.y;
  if (e >= Etot) return;
  int src, dst; edge_pair(ei, e, E, src, dst);
  const float4 l4 = *(const float4*)(xl + ((size_t)b * N + src) * DW + lane * 4);
  const float4 r4 = *(const float4*)(xr + ((size_t)b * N + dst) * DW + lane * 4);
  const float4 a4 = *(const float4*)(att + lane * 4);
  float s = 0.f, t;
  t = l4.x + r4.x; t = t > 0.f ? t : NEG_SLOPE * t; s += t * a4.x;
  t = l4.y + r4.y; t = t > 0.f ? t : NEG_SLOPE * t; s += t * a4.y;
  t = l4.z + r4.z; t = t > 0.f ? t : NEG_SLOPE * t; s += t * a4.z;
  t = l4.w + r4.w; t = t > 0.f ? t : NEG_SLOPE * t; s += t * a4.w;
  s += __shfl_xor(s, 1, 32);
  s += __shfl_xor(s, 2, 32);
  s += __shfl_xor(s, 4, 32);
  s += __shfl_xor(s, 8, 32);                 // lanes 0..15: head0 sum; 16..31: head1 sum
  if (lane == 0 || lane == 16) {
    const int h = lane >> 4;
    scores[((size_t)b * Etot + e) * HH + h] = s;
    atomicMaxF32(&mbuf[((size_t)b * N + dst) * HH + h], s);
  }
}

// thread per (b, edge, head): exp(score - max) and denom accumulation
__global__ void edge_exp_kernel(const int* __restrict__ ei, float* __restrict__ scores,
                                const float* __restrict__ mbuf, float* __restrict__ denom,
                                int E, int Etot, int N) {
  const int i = blockIdx.x * blockDim.x + threadIdx.x;
  if (i >= BB * Etot * HH) return;
  const int h = i & 1;
  const int e = (i >> 1) % Etot;
  const int b = i / (Etot * HH);
  int src, dst; edge_pair(ei, e, E, src, dst);
  float mm = mbuf[((size_t)b * N + dst) * HH + h];
  if (mm < -1e37f) mm = 0.f;                  // jnp.where(isfinite(m), m, 0)
  const float ex = __expf(scores[i] - mm);
  scores[i] = ex;
  atomicAdd(&denom[((size_t)b * N + dst) * HH + h], ex);
}

// one wave per (edge, batch): scatter alpha * xl[src] into agg[dst]
__global__ void __launch_bounds__(256)
aggregate_kernel(const float* __restrict__ xl, const int* __restrict__ ei,
                 const float* __restrict__ scores, const float* __restrict__ denom,
                 float* __restrict__ agg, int E, int Etot, int N) {
  const int e    = (blockIdx.x * 256 + threadIdx.x) >> 5;
  const int lane = threadIdx.x & 31;
  const int b    = blockIdx.y;
  if (e >= Etot) return;
  int src, dst; edge_pair(ei, e, E, src, dst);
  const int h = lane >> 4;                    // lanes 0-15 -> head0 channels, 16-31 -> head1
  const float alpha = scores[((size_t)b * Etot + e) * HH + h] /
                      (denom[((size_t)b * N + dst) * HH + h] + 1e-16f);
  const float4 l4 = *(const float4*)(xl + ((size_t)b * N + src) * DW + lane * 4);
  float* ap = agg + ((size_t)b * N + dst) * DW + lane * 4;
  atomicAdd(ap + 0, alpha * l4.x);
  atomicAdd(ap + 1, alpha * l4.y);
  atomicAdd(ap + 2, alpha * l4.z);
  atomicAdd(ap + 3, alpha * l4.w);
}

// h1 = agg + bias, re-encoded f16 as layer-2 GEMM input
__global__ void finalize_h1_kernel(const float* __restrict__ agg, const float* __restrict__ bias,
                                   _Float16* __restrict__ h1H, int N) {
  const int i = blockIdx.x * blockDim.x + threadIdx.x;  // B*N*DW
  if (i >= BB * N * DW) return;
  h1H[i] = (_Float16)(agg[i] + bias[i & (DW - 1)]);
}

// out[b, j, n] = agg[(b*N+n)*DW + j] + bias[j]
__global__ void finalize_out_kernel(const float* __restrict__ agg, const float* __restrict__ bias,
                                    float* __restrict__ out, int N) {
  const int i = blockIdx.x * blockDim.x + threadIdx.x;  // B*N*DW
  if (i >= BB * N * DW) return;
  const int j = i & (DW - 1);
  const int n = (i >> 7) % N;
  const int b = (i >> 7) / N;
  out[((size_t)b * DW + j) * N + n] = agg[i] + bias[j];
}

// ---------------- launch ----------------
extern "C" void kernel_launch(void* const* d_in, const int* in_sizes, int n_in,
                              void* d_out, int out_size, void* d_ws, size_t ws_size,
                              hipStream_t stream) {
  const float* x     = (const float*)d_in[0];
  const int*   ei    = (const int*)d_in[1];
  const float* Wl1   = (const float*)d_in[2];
  const float* bl1   = (const float*)d_in[3];
  const float* Wr1   = (const float*)d_in[4];
  const float* br1   = (const float*)d_in[5];
  const float* att1  = (const float*)d_in[6];
  const float* bias1 = (const float*)d_in[7];
  const float* Wl2   = (const float*)d_in[8];
  const float* bl2   = (const float*)d_in[9];
  const float* Wr2   = (const float*)d_in[10];
  const float* br2   = (const float*)d_in[11];
  const float* att2  = (const float*)d_in[12];
  const float* bias2 = (const float*)d_in[13];
  float* out = (float*)d_out;

  const int N      = in_sizes[0] / (BB * FIN);
  const int E      = in_sizes[1] / 2;
  const int Etot   = E + N;                    // with self-loops
  const int rows   = BB * N;
  const int rows16 = (rows + 15) & ~15;        // padded for guard-free WMMA tiles

  // carve scratch from d_ws (256B-aligned slices); row-padded buffers
  size_t off = 0; char* base = (char*)d_ws;
  auto alloc = [&](size_t bytes) -> void* {
    void* p = base + off;
    off += (bytes + 255) & ~(size_t)255;
    return p;
  };
  _Float16* featH  = (_Float16*)alloc((size_t)rows16 * FIN * sizeof(_Float16));
  _Float16* h1H    = (_Float16*)alloc((size_t)rows16 * DW * sizeof(_Float16));
  float*    xl     = (float*)alloc((size_t)rows16 * DW * sizeof(float));
  float*    xr     = (float*)alloc((size_t)rows16 * DW * sizeof(float));
  float*    scores = (float*)alloc((size_t)BB * Etot * HH * sizeof(float));
  float*    mbuf   = (float*)alloc((size_t)rows * HH * sizeof(float));
  float*    denom  = (float*)alloc((size_t)rows * HH * sizeof(float));
  float*    agg    = (float*)alloc((size_t)rows * DW * sizeof(float));
  (void)ws_size; (void)n_in; (void)out_size;

  const int T = 256;
  const int gblocks = (rows16 / 16 + 7) / 8;   // 8 waves (row-tiles) per block
  const int eblocks = (Etot + 7) / 8;          // 8 waves (edges) per block
  const dim3 egrid(eblocks, BB);

  pack_feat_kernel<<<(rows * FIN + T - 1) / T, T, 0, stream>>>(x, featH, N);

  auto run_edges = [&](const float* att, const float* bias, bool last) {
    fill_kernel<<<(rows * HH + T - 1) / T, T, 0, stream>>>(mbuf, -3.0e38f, rows * HH);
    fill_kernel<<<(rows * HH + T - 1) / T, T, 0, stream>>>(denom, 0.f, rows * HH);
    fill_kernel<<<(rows * DW + T - 1) / T, T, 0, stream>>>(agg, 0.f, rows * DW);
    edge_score_kernel<<<egrid, 256, 0, stream>>>(xl, xr, ei, att, scores, mbuf, E, Etot, N);
    edge_exp_kernel<<<(BB * Etot * HH + T - 1) / T, T, 0, stream>>>(ei, scores, mbuf, denom, E, Etot, N);
    aggregate_kernel<<<egrid, 256, 0, stream>>>(xl, ei, scores, denom, agg, E, Etot, N);
    if (!last)
      finalize_h1_kernel<<<(rows * DW + T - 1) / T, T, 0, stream>>>(agg, bias, h1H, N);
    else
      finalize_out_kernel<<<(rows * DW + T - 1) / T, T, 0, stream>>>(agg, bias, out, N);
  };

  // layer 1
  gemm_wmma_kernel<FIN><<<gblocks, 256, 0, stream>>>(featH, Wl1, bl1, xl, rows16);
  gemm_wmma_kernel<FIN><<<gblocks, 256, 0, stream>>>(featH, Wr1, br1, xr, rows16);
  run_edges(att1, bias1, false);

  // layer 2
  gemm_wmma_kernel<DW><<<gblocks, 256, 0, stream>>>(h1H, Wl2, bl2, xl, rows16);
  gemm_wmma_kernel<DW><<<gblocks, 256, 0, stream>>>(h1H, Wr2, br2, xr, rows16);
  run_edges(att2, bias2, true);
}
